// Wadain_35038343201604
// MI455X (gfx1250) — matrix operand order repaired and checked by reading, not compiled, not run on hardware
//
#include <hip/hip_runtime.h>
#include <hip/hip_bf16.h>

// Problem constants (from reference)
#define B_    16
#define CIN_  512
#define COUT_ 512
#define K_    3
#define T_    2048
#define S_    128
#define TPAD_ (T_ + 2)   // 2050: one zero halo row each side of t

typedef __attribute__((ext_vector_type(16))) __bf16        v16bf;
typedef __attribute__((ext_vector_type(8)))  float         v8f;
typedef __attribute__((ext_vector_type(4)))  unsigned int  v4u;
typedef __attribute__((ext_vector_type(8)))  int           v8i;
typedef __attribute__((ext_vector_type(4)))  int           v4i;

union Frag16 { v16bf v; v4u q[2]; };  // one 32B bf16 WMMA A/B fragment per lane

// LDS tile geometry: rows of 32 halfs padded by TDM to 40 halfs (80B pitch:
// 16B-aligned for ds_load_b128, 20-bank stride -> conflict-free over 16 lanes).
#define PITCH  40
#define XT_SZ  (130 * PITCH)   // activation tile: 130 t-cols x 32 ci
#define WT_SZ  (128 * PITCH)   // weight tile:     128 co      x 32 ci

// ---------------------------------------------------------------------------
// TDM 2D tile load: global (row-major, stride0 elements of 2B) -> LDS with
// padding (every 16 DWORDs stored, insert 4 DWORDs) => 40-half LDS pitch.
// Descriptor bitfields per CDNA5 ISA §8.3/8.4 (count=1, type=2, data_size=2B).
// ---------------------------------------------------------------------------
__device__ inline void tdm_load_2d(unsigned lds_byte_off,
                                   const void* gptr,
                                   unsigned tensor_d0, unsigned tensor_d1,
                                   unsigned tile_d0,   unsigned tile_d1,
                                   unsigned stride0) {
    unsigned long long ga = (unsigned long long)gptr;
    v4u g0;
    g0[0] = 1u;                                   // count=1 (valid user D#)
    g0[1] = lds_byte_off;                         // lds_addr (bytes)
    g0[2] = (unsigned)(ga & 0xffffffffu);         // global_addr[31:0]
    g0[3] = (unsigned)((ga >> 32) & 0x1ffffffu)   // global_addr[56:32]
            | (2u << 30);                         // type=2 ("image")
    v8i g1;
    g1[0] = (int)((1u << 16)      // data_size = 1 -> 2 bytes
                | (1u << 20)      // pad_enable
                | (3u << 22)      // pad_interval code 3 -> every 16 DWORDs
                | (3u << 25));    // pad_amount  code 3 -> insert 4 DWORDs
    g1[1] = (int)(tensor_d0 << 16);                         // dim0[15:0]
    g1[2] = (int)((tensor_d0 >> 16) | (tensor_d1 << 16));   // dim0 hi | dim1 lo
    g1[3] = (int)((tensor_d1 >> 16) | (tile_d0  << 16));    // dim1 hi | tile_dim0
    g1[4] = (int)(tile_d1 & 0xffff);                        // tile_dim1 (tile_dim2=0)
    g1[5] = (int)stride0;                                   // dim0_stride[31:0]
    g1[6] = 0;                                              // stride hi / dim1_stride
    g1[7] = 0;
    v4i z4 = {0, 0, 0, 0};
#if __clang_major__ >= 23
    v8i z8 = {0, 0, 0, 0, 0, 0, 0, 0};
    __builtin_amdgcn_tensor_load_to_lds(g0, g1, z4, z4, z8, 0);
#else
    __builtin_amdgcn_tensor_load_to_lds(g0, g1, z4, z4, 0);
#endif
}

// ---------------------------------------------------------------------------
// Kernel 1: s[b,ci] = ls * dot(c_trg[b,:], style_w[ci,:]) + style_b[ci]
// ---------------------------------------------------------------------------
__global__ void style_kernel(const float* __restrict__ c_trg,
                             const float* __restrict__ style_w,
                             const float* __restrict__ style_b,
                             float* __restrict__ s_out) {
    int idx = blockIdx.x * blockDim.x + threadIdx.x;
    int b  = idx >> 9;
    int ci = idx & (CIN_ - 1);
    const float ls = 0.08838834764831845f;   // 1/sqrt(128)
    const float* ct = c_trg   + (size_t)b  * S_;
    const float* sw = style_w + (size_t)ci * S_;
    float acc = 0.0f;
#pragma unroll 8
    for (int i = 0; i < S_; ++i) acc += ct[i] * sw[i];
    s_out[idx] = acc * ls + style_b[ci];
}

// ---------------------------------------------------------------------------
// Kernel 2: per (b,co) wave: modulate, demod, mean-subtract, store bf16
// weights in layout W[b][k][co][ci].
// ---------------------------------------------------------------------------
__global__ __launch_bounds__(256)
void modw_kernel(const float* __restrict__ s,
                 const float* __restrict__ conv_w,
                 __bf16* __restrict__ W) {
    const int wave = threadIdx.x >> 5;
    const int lane = threadIdx.x & 31;
    const int gw   = blockIdx.x * 8 + wave;
    const int b    = gw >> 9;
    const int co   = gw & (COUT_ - 1);
    const float cs = 0.014731391274719738f;  // 1/sqrt(CIN_*K_*K_)

    float sum = 0.0f, ssq = 0.0f;
    for (int ci = lane; ci < CIN_; ci += 32) {
        float sv = s[b * CIN_ + ci];
#pragma unroll
        for (int k = 0; k < K_; ++k) {
            float cw = conv_w[((size_t)co * CIN_ + ci) * K_ + k];
            float w  = cs * (sv * cw + sv);
            sum += w;
            ssq += w * w;
        }
    }
#pragma unroll
    for (int m = 16; m >= 1; m >>= 1) {      // wave32 butterfly
        sum += __shfl_xor(sum, m, 32);
        ssq += __shfl_xor(ssq, m, 32);
    }
    const float mean  = sum * (1.0f / (float)(CIN_ * K_));
    const float demod = rsqrtf(ssq + 1e-8f);

    for (int ci = lane; ci < CIN_; ci += 32) {
        float sv = s[b * CIN_ + ci];
#pragma unroll
        for (int k = 0; k < K_; ++k) {
            float cw = conv_w[((size_t)co * CIN_ + ci) * K_ + k];
            float w  = cs * (sv * cw + sv);
            W[(((size_t)(b * K_ + k) * COUT_) + co) * CIN_ + ci] =
                (__bf16)((w - mean) * demod);
        }
    }
}

// ---------------------------------------------------------------------------
// Kernel P1: transpose+convert x[b][ci][t] fp32 -> XTg[b][t+1][ci] bf16.
// 64x64 tiles through LDS; both global phases coalesced.
// ---------------------------------------------------------------------------
__global__ __launch_bounds__(256)
void xpose_kernel(const float* __restrict__ x, __bf16* __restrict__ XTg) {
    __shared__ float tile[64][65];
    const int b   = blockIdx.z;
    const int ci0 = blockIdx.y * 64;
    const int t0  = blockIdx.x * 64;
    const int tid = threadIdx.x;

    for (int i = tid; i < 64 * 64; i += 256) {
        int r = i >> 6, c = i & 63;              // r: ci, c: t (coalesced read)
        tile[r][c] = x[((size_t)(b * CIN_ + ci0 + r)) * T_ + t0 + c];
    }
    __syncthreads();
    for (int i = tid; i < 64 * 64; i += 256) {
        int r = i >> 6, c = i & 63;              // r: t, c: ci (coalesced write)
        XTg[((size_t)(b * TPAD_ + t0 + r + 1)) * CIN_ + ci0 + c] =
            (__bf16)tile[c][r];
    }
}

// Kernel P2: zero the two halo rows (t = -1 and t = T_) of XTg.
__global__ void zpad_kernel(__bf16* __restrict__ XTg) {
    int idx = blockIdx.x * blockDim.x + threadIdx.x;  // 0 .. 16*2*512-1
    int b   = idx >> 10;
    int row = ((idx >> 9) & 1) ? (TPAD_ - 1) : 0;
    int ci  = idx & (CIN_ - 1);
    XTg[((size_t)(b * TPAD_ + row)) * CIN_ + ci] = (__bf16)0.0f;
}

// ---------------------------------------------------------------------------
// Kernel 3: implicit-GEMM conv via v_wmma_f32_16x16x32_bf16, all tiles fed
// by the Tensor Data Mover (global->LDS DMA, TENSORcnt-tracked, LDS-padded).
// Block: 256 threads = 8 waves; tile 128(co) x 128(t).
// Wave (wm,wn): 64co x 32t = 4x2 C-tiles. K loop: ci in steps of 32.
// ---------------------------------------------------------------------------
__global__ __launch_bounds__(256)
void conv_wmma_kernel(const __bf16* __restrict__ XTg,
                      const __bf16* __restrict__ W,
                      float* __restrict__ out) {
    __shared__ __attribute__((aligned(16))) __bf16 smem[XT_SZ + 3 * WT_SZ];

    const int b      = blockIdx.z;
    const int co_blk = blockIdx.y * 128;
    const int t_blk  = blockIdx.x * 128;
    const int wave   = threadIdx.x >> 5;
    const int lane   = threadIdx.x & 31;
    const int wm     = wave & 1;        // 0..1 : 64-row half
    const int wn     = wave >> 1;       // 0..3 : 32-col quarter
    const int laneLo = lane & 15;
    const int laneHi = lane >> 4;

    const unsigned lds_base = (unsigned)(size_t)(&smem[0]);

    v8f acc[4][2];
    const v8f vzero = {};
#pragma unroll
    for (int ms = 0; ms < 4; ++ms)
#pragma unroll
        for (int ns = 0; ns < 2; ++ns) acc[ms][ns] = vzero;

    for (int ci0 = 0; ci0 < CIN_; ci0 += 32) {
        __syncthreads();   // previous step's fragment reads complete
        if (wave == 0) {
            // Activation tile: rows tpad = t_blk .. t_blk+129, cols ci0..ci0+31.
            tdm_load_2d(lds_base,
                        XTg + ((size_t)(b * TPAD_ + t_blk)) * CIN_ + ci0,
                        (unsigned)(CIN_ - ci0), (unsigned)(TPAD_ - t_blk),
                        32u, 130u, (unsigned)CIN_);
            // Weight tiles, one per tap k: rows co_blk..+127, cols ci0..+31.
#pragma unroll
            for (int k = 0; k < K_; ++k)
                tdm_load_2d(lds_base + (unsigned)(XT_SZ + k * WT_SZ) * 2u,
                            W + (((size_t)(b * K_ + k) * COUT_) + co_blk) * CIN_ + ci0,
                            (unsigned)(CIN_ - ci0), (unsigned)(COUT_ - co_blk),
                            32u, 128u, (unsigned)CIN_);
        }
        __builtin_amdgcn_s_wait_tensorcnt(0);
        __syncthreads();

#pragma unroll
        for (int k = 0; k < K_; ++k) {
            // B fragments (32x16): lane N=laneLo, K rows laneHi*16..+15 are
            // 16 contiguous halfs in the activation tile row (ci-contiguous).
            Frag16 bf[2];
#pragma unroll
            for (int ns = 0; ns < 2; ++ns) {
                int c = wn * 32 + ns * 16 + laneLo + k;  // <= 129
                const v4u* p =
                    reinterpret_cast<const v4u*>(&smem[c * PITCH + laneHi * 16]);
                bf[ns].q[0] = p[0];
                bf[ns].q[1] = p[1];
            }
#pragma unroll
            for (int ms = 0; ms < 4; ++ms) {
                // A fragment (16x32): lanes 0-15 hold K [0..15],
                // lanes 16-31 hold K [16..31].
                const __bf16* wrow =
                    &smem[XT_SZ + k * WT_SZ +
                          (wm * 64 + ms * 16 + laneLo) * PITCH];
                Frag16 af;
                af.q[0] = *reinterpret_cast<const v4u*>(wrow + laneHi * 16);
                af.q[1] = *reinterpret_cast<const v4u*>(wrow + laneHi * 16 + 8);
#pragma unroll
                for (int ns = 0; ns < 2; ++ns)
                    acc[ms][ns] = __builtin_amdgcn_wmma_f32_16x16x32_bf16(
                        false, af.v, false, bf[ns].v,
                        (short)0, acc[ms][ns], false, false);
            }
        }
    }

    // Epilogue: C layout — VGPR j: lanes 0-15 -> M=j, lanes 16-31 -> M=j+8.
#pragma unroll
    for (int ms = 0; ms < 4; ++ms) {
#pragma unroll
        for (int ns = 0; ns < 2; ++ns) {
            int col = t_blk + wn * 32 + ns * 16 + laneLo;
#pragma unroll
            for (int j = 0; j < 8; ++j) {
                int row = co_blk + wm * 64 + ms * 16 + laneHi * 8 + j;
                out[((size_t)(b * COUT_ + row)) * T_ + col] = acc[ms][ns][j];
            }
        }
    }
}

// ---------------------------------------------------------------------------
extern "C" void kernel_launch(void* const* d_in, const int* in_sizes, int n_in,
                              void* d_out, int out_size, void* d_ws, size_t ws_size,
                              hipStream_t stream) {
    const float* x       = (const float*)d_in[0];  // (B, CIN, T)
    const float* c_trg   = (const float*)d_in[1];  // (B, S)
    const float* style_w = (const float*)d_in[2];  // (CIN, S)
    const float* style_b = (const float*)d_in[3];  // (CIN,)
    const float* conv_w  = (const float*)d_in[4];  // (1, COUT, CIN, K)
    float* out = (float*)d_out;                    // (B, COUT, T) fp32

    // Workspace layout:
    //   [0, 32KB)        : s fp32 (B*CIN)
    //   [32KB, +24MB)    : W bf16 [B][K][COUT][CIN]
    //   [.., +32MB)      : XTg bf16 [B][TPAD_][CIN] (transposed, halo-padded x)
    float*  s_buf = (float*)d_ws;
    __bf16* Wbuf  = (__bf16*)((char*)d_ws + 32768);
    __bf16* XTg   = (__bf16*)((char*)d_ws + 32768 +
                              (size_t)B_ * K_ * COUT_ * CIN_ * 2);

    style_kernel<<<(B_ * CIN_) / 256, 256, 0, stream>>>(c_trg, style_w, style_b, s_buf);
    modw_kernel<<<(B_ * COUT_) / 8, 256, 0, stream>>>(s_buf, conv_w, Wbuf);

    dim3 gx(T_ / 64, CIN_ / 64, B_);
    xpose_kernel<<<gx, 256, 0, stream>>>(x, XTg);
    zpad_kernel<<<(B_ * 2 * CIN_) / 256, 256, 0, stream>>>(XTg);

    dim3 grid(T_ / 128, COUT_ / 128, B_);
    conv_wmma_kernel<<<grid, 256, 0, stream>>>(XTg, Wbuf, out);
}
